// enformer_performer_55765855371712
// MI455X (gfx1250) — compile-verified, matrix operand order checked by reading, not run on hardware
//
#include <hip/hip_runtime.h>
#include <hip/hip_bf16.h>

typedef __attribute__((ext_vector_type(16))) __bf16 v16bf;
typedef __attribute__((ext_vector_type(8)))  float  v8f;

#define BN_SCALE 0.9995003746f   // 1/sqrt(1 + 0.001)
#define STABC    1e-3f
#define LN10000  9.210340371976184f
#define BT_W     40              // padded LDS row (bank-conflict-free, 16B aligned)

__device__ __forceinline__ unsigned short f2bfu(float f) {
  unsigned u = __builtin_bit_cast(unsigned, f);
  return (unsigned short)((u + 0x7FFFu + ((u >> 16) & 1u)) >> 16);
}
__device__ __forceinline__ __bf16 f2bf(float f) {
  unsigned short s = f2bfu(f);
  return __builtin_bit_cast(__bf16, s);
}
__device__ __forceinline__ float gelu_f(float x) {
  return 0.5f * x * (1.f + tanhf(0.7978845608028654f * (x + 0.044715f * x * x * x)));
}
__device__ __forceinline__ float act_f(float v, int act) {
  if (act == 1) return gelu_f(v);
  if (act == 2) return fmaxf(v, 0.f);
  if (act == 3) return (v > 20.f) ? v : log1pf(expf(v));
  return v;
}

// ---- B-tile staging: 32(K) x 32(N) -> LDS bf16, transposed bt[n][k] ----
// Interior: fully unconditional. Each thread: 8 strided loads along k at one
// column (coalesced along n across lanes), one packed ds_store_b128.
__device__ __forceinline__ void stage_b_full(__bf16 (*bt)[BT_W], const float* __restrict__ pB,
                                             int tid, long b_ks, long b_ns)
{
  int tn = tid & 31;
  int tk = (tid >> 5) << 3;             // 0,8,16,24
  const float* p = pB + (long)tk * b_ks + (long)tn * b_ns;
  unsigned short r[8];
#pragma unroll
  for (int j = 0; j < 8; ++j) r[j] = f2bfu(p[j * b_ks]);
  uint4 u;
  u.x = (unsigned)r[0] | ((unsigned)r[1] << 16);
  u.y = (unsigned)r[2] | ((unsigned)r[3] << 16);
  u.z = (unsigned)r[4] | ((unsigned)r[5] << 16);
  u.w = (unsigned)r[6] | ((unsigned)r[7] << 16);
  *(uint4*)(&bt[tn][tk]) = u;
}

// Edge: clamped + zero-filled (only last n-block / K tail blocks take this).
__device__ __forceinline__ void stage_b_edge(__bf16 (*bt)[BT_W], const float* __restrict__ Bb,
                                             int tid, int k0, int n0, int K, int N,
                                             long b_ks, long b_ns)
{
  int tn = tid & 31;
  int tk = (tid >> 5) << 3;
  int ng = n0 + tn;
  int nc = ng < N ? ng : N - 1;
  unsigned short r[8];
#pragma unroll
  for (int j = 0; j < 8; ++j) {
    int kg = k0 + tk + j;
    int kc = kg < K ? kg : K - 1;
    float v = Bb[(long)kc * b_ks + (long)nc * b_ns];
    r[j] = (kg < K && ng < N) ? f2bfu(v) : (unsigned short)0;
  }
  uint4 u;
  u.x = (unsigned)r[0] | ((unsigned)r[1] << 16);
  u.y = (unsigned)r[2] | ((unsigned)r[3] << 16);
  u.z = (unsigned)r[4] | ((unsigned)r[5] << 16);
  u.w = (unsigned)r[6] | ((unsigned)r[7] << 16);
  *(uint4*)(&bt[tn][tk]) = u;
}

// Read a lane's 16x32-bf16 B fragment (two ds_load_b128) for n-half `half`.
__device__ __forceinline__ v16bf load_bfrag(__bf16 (*bt)[BT_W], int lane, int base8, int half)
{
  union { uint4 q[2]; v16bf v; } ub;
  const __bf16* brow = &bt[(lane & 15) + half][0];
  ub.q[0] = *(const uint4*)(brow + base8);
  ub.q[1] = *(const uint4*)(brow + base8 + 16);
  return ub.v;
}

// ---------------------------------------------------------------------------
// Generic strided GEMM: Out[b,m,n] = act(ascale*A[b,m,:] . B[b,:,n] + bias[n] + Res[b,m,n])
// Block = 64(M) x 32(N): 4 waves share one 32-col B tile via LDS; each wave
// owns a 16x32 tile (one A fragment feeds two independent WMMAs per chunk).
// ---------------------------------------------------------------------------
__global__ void gemm_kernel(const float* __restrict__ A, const float* __restrict__ B,
                            const float* __restrict__ bias, const float* __restrict__ Res,
                            float* __restrict__ Out,
                            int M, int N, int K,
                            long a_rs, long a_ks, long a_bs,
                            long b_ks, long b_ns, long b_bs,
                            long o_rs, long o_bs,
                            long r_rs, long r_bs,
                            float ascale, int act)
{
  __shared__ __bf16 bt[32][BT_W];
  int tid = threadIdx.x;
  int lane = tid & 31;
  int wv = tid >> 5;
  int m0 = (blockIdx.x * 4 + wv) * 16;
  int n0 = blockIdx.y * 32;
  int bz = blockIdx.z;
  const float* Ab = A + (long)bz * a_bs;
  const float* Bb = B + (long)bz * b_bs;

  int arow = m0 + (lane & 15);
  int arow_c = arow < M ? arow : M - 1;           // clamped (stores masked later)
  int base8 = (lane & 16) >> 1;                    // 0 or 8
  const float* Arow = Ab + (long)arow_c * a_rs;
  bool fastA = (a_ks == 1) && ((a_rs & 3) == 0);
  bool nfull = (n0 + 32 <= N);
  const float* pB = Bb + (long)n0 * b_ns;          // advanced by 32*b_ks per chunk

  long aoff[16];
#pragma unroll
  for (int i = 0; i < 16; ++i)
    aoff[i] = (long)(((i & 7) | ((i & 8) << 1)) + base8) * a_ks;

  v8f acc[2];
  acc[0] = (v8f){0.f, 0.f, 0.f, 0.f, 0.f, 0.f, 0.f, 0.f};
  acc[1] = (v8f){0.f, 0.f, 0.f, 0.f, 0.f, 0.f, 0.f, 0.f};

  int k0 = 0;
  for (; k0 + 32 <= K; k0 += 32, pB += 32 * b_ks) {
    __syncthreads();
    if (nfull) stage_b_full(bt, pB, tid, b_ks, b_ns);
    else       stage_b_edge(bt, Bb, tid, k0, n0, K, N, b_ks, b_ns);
    __syncthreads();
    v16bf af;
    if (fastA) {
      const float* p = Arow + k0 + base8;
      float4 f0 = *(const float4*)(p);
      float4 f1 = *(const float4*)(p + 4);
      float4 f2 = *(const float4*)(p + 16);
      float4 f3 = *(const float4*)(p + 20);
      af[0]  = f2bf(f0.x * ascale); af[1]  = f2bf(f0.y * ascale);
      af[2]  = f2bf(f0.z * ascale); af[3]  = f2bf(f0.w * ascale);
      af[4]  = f2bf(f1.x * ascale); af[5]  = f2bf(f1.y * ascale);
      af[6]  = f2bf(f1.z * ascale); af[7]  = f2bf(f1.w * ascale);
      af[8]  = f2bf(f2.x * ascale); af[9]  = f2bf(f2.y * ascale);
      af[10] = f2bf(f2.z * ascale); af[11] = f2bf(f2.w * ascale);
      af[12] = f2bf(f3.x * ascale); af[13] = f2bf(f3.y * ascale);
      af[14] = f2bf(f3.z * ascale); af[15] = f2bf(f3.w * ascale);
    } else {
      const float* p = Arow + (long)k0 * a_ks;
#pragma unroll
      for (int i = 0; i < 16; ++i) af[i] = f2bf(p[aoff[i]] * ascale);
    }
    v16bf b0 = load_bfrag(bt, lane, base8, 0);
    v16bf b1 = load_bfrag(bt, lane, base8, 16);
    acc[0] = __builtin_amdgcn_wmma_f32_16x16x32_bf16(false, af, false, b0, (short)0, acc[0], false, false);
    acc[1] = __builtin_amdgcn_wmma_f32_16x16x32_bf16(false, af, false, b1, (short)0, acc[1], false, false);
  }
  if (k0 < K) {           // K tail, clamped + masked
    __syncthreads();
    stage_b_edge(bt, Bb, tid, k0, n0, K, N, b_ks, b_ns);
    __syncthreads();
    v16bf af;
#pragma unroll
    for (int i = 0; i < 16; ++i) {
      int k = k0 + ((i & 7) | ((i & 8) << 1)) + base8;
      int kc = k < K ? k : K - 1;
      float av = Arow[(long)kc * a_ks];
      af[i] = f2bf(k < K ? av * ascale : 0.f);
    }
    v16bf b0 = load_bfrag(bt, lane, base8, 0);
    v16bf b1 = load_bfrag(bt, lane, base8, 16);
    acc[0] = __builtin_amdgcn_wmma_f32_16x16x32_bf16(false, af, false, b0, (short)0, acc[0], false, false);
    acc[1] = __builtin_amdgcn_wmma_f32_16x16x32_bf16(false, af, false, b1, (short)0, acc[1], false, false);
  }

#pragma unroll
  for (int t = 0; t < 2; ++t) {
    int n = n0 + 16 * t + (lane & 15);
    if (n >= N) continue;
    float bi = bias ? bias[n] : 0.f;
#pragma unroll
    for (int j = 0; j < 8; ++j) {
      int m = m0 + j + base8;
      if (m >= M) continue;
      float v = acc[t][j] + bi;
      if (Res) v += Res[(long)bz * r_bs + (long)m * r_rs + n];
      Out[(long)bz * o_bs + (long)m * o_rs + n] = act_f(v, act);
    }
  }
}

// ---------------------------------------------------------------------------
// Conv1d ('SAME', stride 1, odd W) as implicit im2col GEMM on WMMA.
// Fuses y = gelu(bn_g*x*BN_SCALE + bn_b) on the input (conv_block), plus
// bias / residual / epilogue activation. Fast path when Cin % 32 == 0.
// ---------------------------------------------------------------------------
__global__ void conv1d_kernel(const float* __restrict__ X, const float* __restrict__ Wt,
                              const float* __restrict__ bias,
                              const float* __restrict__ bn_g, const float* __restrict__ bn_b,
                              const float* __restrict__ Res,
                              float* __restrict__ Out,
                              int L, int Cin, int Cout, int W, int act)
{
  __shared__ __bf16 bt[32][BT_W];
  int tid = threadIdx.x;
  int lane = tid & 31;
  int wv = tid >> 5;
  int m0 = (blockIdx.x * 4 + wv) * 16;
  int n0 = blockIdx.y * 32;
  int K = W * Cin;
  int pad = W >> 1;
  int arow = m0 + (lane & 15);
  int arow_c = arow < L ? arow : L - 1;
  int base8 = (lane & 16) >> 1;
  bool fast = (Cin & 31) == 0;
  bool nfull = (n0 + 32 <= Cout);
  const float* pW = Wt + n0;                       // advanced by 32*Cout per chunk

  v8f acc[2];
  acc[0] = (v8f){0.f, 0.f, 0.f, 0.f, 0.f, 0.f, 0.f, 0.f};
  acc[1] = (v8f){0.f, 0.f, 0.f, 0.f, 0.f, 0.f, 0.f, 0.f};

  for (int k0 = 0; k0 < K; k0 += 32, pW += 32 * (long)Cout) {
    __syncthreads();
    bool kfull = (k0 + 32 <= K);
    if (nfull && kfull) stage_b_full(bt, pW, tid, (long)Cout, 1);
    else                stage_b_edge(bt, Wt, tid, k0, n0, K, Cout, (long)Cout, 1);
    __syncthreads();
    v16bf af;
    if (fast) {
      int w = k0 / Cin;                 // whole chunk inside one tap
      int c0 = k0 - w * Cin;
      int p = arow_c + w - pad;
      bool valid = (arow < L) && (p >= 0) && (p < L);
      int pc = p < 0 ? 0 : (p >= L ? L - 1 : p);
      const float* xp = X + (long)pc * Cin + c0 + base8;
      float4 f0 = *(const float4*)(xp);
      float4 f1 = *(const float4*)(xp + 4);
      float4 f2v = *(const float4*)(xp + 16);
      float4 f3 = *(const float4*)(xp + 20);
      float t[16] = {f0.x, f0.y, f0.z, f0.w, f1.x, f1.y, f1.z, f1.w,
                     f2v.x, f2v.y, f2v.z, f2v.w, f3.x, f3.y, f3.z, f3.w};
#pragma unroll
      for (int i = 0; i < 16; ++i) {
        int c = c0 + base8 + ((i & 7) | ((i & 8) << 1));
        float xv = t[i];
        if (bn_g) xv = gelu_f(bn_g[c] * xv * BN_SCALE + bn_b[c]);
        af[i] = f2bf(valid ? xv : 0.f);
      }
    } else {
#pragma unroll
      for (int i = 0; i < 16; ++i) {
        int k = k0 + ((i & 7) | ((i & 8) << 1)) + base8;
        int kc = k < K ? k : K - 1;
        int w = kc / Cin, c = kc - w * Cin;
        int p = arow_c + w - pad;
        bool valid = (k < K) && (arow < L) && (p >= 0) && (p < L);
        int pc = p < 0 ? 0 : (p >= L ? L - 1 : p);
        float xv = X[(long)pc * Cin + c];
        if (bn_g) xv = gelu_f(bn_g[c] * xv * BN_SCALE + bn_b[c]);
        af[i] = f2bf(valid ? xv : 0.f);
      }
    }
    v16bf b0 = load_bfrag(bt, lane, base8, 0);
    v16bf b1 = load_bfrag(bt, lane, base8, 16);
    acc[0] = __builtin_amdgcn_wmma_f32_16x16x32_bf16(false, af, false, b0, (short)0, acc[0], false, false);
    acc[1] = __builtin_amdgcn_wmma_f32_16x16x32_bf16(false, af, false, b1, (short)0, acc[1], false, false);
  }

#pragma unroll
  for (int t = 0; t < 2; ++t) {
    int n = n0 + 16 * t + (lane & 15);
    if (n >= Cout) continue;
    float bi = bias ? bias[n] : 0.f;
#pragma unroll
    for (int j = 0; j < 8; ++j) {
      int m = m0 + j + base8;
      if (m >= L) continue;
      float v = acc[t][j] + bi;
      if (Res) v += Res[(long)m * Cout + n];
      Out[(long)m * Cout + n] = act_f(v, act);
    }
  }
}

// ---------------------------------------------------------------------------
// LayerNorm over last dim (one block per row)
// ---------------------------------------------------------------------------
__global__ void ln_kernel(const float* __restrict__ x, const float* __restrict__ g,
                          const float* __restrict__ b, float* __restrict__ y, int D)
{
  int row = blockIdx.x;
  const float* xr = x + (long)row * D;
  float* yr = y + (long)row * D;
  __shared__ float s1[256], s2[256];
  float a = 0.f, q = 0.f;
  for (int i = threadIdx.x; i < D; i += 256) { float v = xr[i]; a += v; q += v * v; }
  s1[threadIdx.x] = a; s2[threadIdx.x] = q; __syncthreads();
  for (int o = 128; o > 0; o >>= 1) {
    if (threadIdx.x < o) { s1[threadIdx.x] += s1[threadIdx.x + o]; s2[threadIdx.x] += s2[threadIdx.x + o]; }
    __syncthreads();
  }
  float mu  = s1[0] / (float)D;
  float var = s2[0] / (float)D - mu * mu;
  float inv = rsqrtf(var + 1e-3f);
  for (int i = threadIdx.x; i < D; i += 256) yr[i] = g[i] * (xr[i] - mu) * inv + b[i];
}

// softmax pooling over pairs: out[l,c] = sum_p softmax_p(logits[2l+p,c]) * x[2l+p,c]
__global__ void pool_kernel(const float* __restrict__ x, const float* __restrict__ lg,
                            float* __restrict__ out, int Lh, int C)
{
  long i = (long)blockIdx.x * blockDim.x + threadIdx.x;
  long tot = (long)Lh * C;
  if (i >= tot) return;
  int c = (int)(i % C); long l = i / C;
  long i0 = (2 * l) * (long)C + c, i1 = i0 + C;
  float l0 = lg[i0], l1 = lg[i1];
  float mx = fmaxf(l0, l1);
  float e0 = expf(l0 - mx), e1 = expf(l1 - mx);
  out[i] = (e0 * x[i0] + e1 * x[i1]) / (e0 + e1);
}

// concat tower + atac along channels, add sinusoidal PE
__global__ void concat_pe_kernel(const float* __restrict__ xt, const float* __restrict__ a,
                                 float* __restrict__ t, int L, int Cx, int Ca)
{
  int C = Cx + Ca;
  long i = (long)blockIdx.x * blockDim.x + threadIdx.x;
  if (i >= (long)L * C) return;
  int c = (int)(i % C); int l = (int)(i / C);
  float v = (c < Cx) ? xt[(long)l * Cx + c] : a[(long)l * Ca + (c - Cx)];
  int ii = c >> 1;
  float ang = (float)l * expf(-(2.f * (float)ii / (float)C) * LN10000);
  v += (c & 1) ? cosf(ang) : sinf(ang);
  t[i] = v;
}

__global__ void rope_kernel(float* __restrict__ x, int L, int H, int Dh)
{
  int half = Dh >> 1;
  long i = (long)blockIdx.x * blockDim.x + threadIdx.x;
  long tot = (long)L * H * half;
  if (i >= tot) return;
  int j = (int)(i % half);
  long lh = i / half;
  int h = (int)(lh % H);
  long l = lh / H;
  float ang = (float)l * expf(-((float)j / (float)half) * LN10000);
  float s = sinf(ang), c = cosf(ang);
  float* p = x + (l * H + h) * (long)Dh;
  float x1 = p[j], x2 = p[j + half];
  p[j]        = x1 * c - x2 * s;
  p[j + half] = x1 * s + x2 * c;
}

__global__ void diag_kernel(const float* __restrict__ q, float* __restrict__ dg,
                            int rows, int D, float dn)
{
  int r = blockIdx.x * blockDim.x + threadIdx.x;
  if (r >= rows) return;
  const float* p = q + (long)r * D;
  float s = 0.f;
  for (int d = 0; d < D; ++d) { float v = p[d] * dn; s += v * v; }
  dg[r] = 0.5f * s;
}

// query features: per-row max over M features  (blockDim == MF == 256)
__global__ void qfeat_kernel(const float* __restrict__ dash, const float* __restrict__ dg,
                             float* __restrict__ qp, int MF_, float ratio)
{
  int r = blockIdx.x;
  __shared__ float s[256];
  float v = dash[(long)r * MF_ + threadIdx.x];
  s[threadIdx.x] = v; __syncthreads();
  for (int o = 128; o > 0; o >>= 1) {
    if (threadIdx.x < o) s[threadIdx.x] = fmaxf(s[threadIdx.x], s[threadIdx.x + o]);
    __syncthreads();
  }
  qp[(long)r * MF_ + threadIdx.x] = ratio * (expf(v - dg[r] - s[0]) + STABC);
}

// key max over (L, M) per head
__global__ void kmax_kernel(const float* __restrict__ dash, float* __restrict__ km,
                            int L, int H, int MF_)
{
  int h = blockIdx.x;
  __shared__ float s[256];
  float mx = -3.4e38f;
  long tot = (long)L * MF_;
  for (long idx = threadIdx.x; idx < tot; idx += 256) {
    long l = idx / MF_; int m = (int)(idx % MF_);
    mx = fmaxf(mx, dash[(l * H + h) * (long)MF_ + m]);
  }
  s[threadIdx.x] = mx; __syncthreads();
  for (int o = 128; o > 0; o >>= 1) {
    if (threadIdx.x < o) s[threadIdx.x] = fmaxf(s[threadIdx.x], s[threadIdx.x + o]);
    __syncthreads();
  }
  if (threadIdx.x == 0) km[h] = s[0];
}

__global__ void kfeat_kernel(const float* __restrict__ dash, const float* __restrict__ dg,
                             const float* __restrict__ km, float* __restrict__ kp,
                             int H, int MF_, long tot, float ratio)
{
  long i = (long)blockIdx.x * blockDim.x + threadIdx.x;
  if (i >= tot) return;
  long r = i / MF_; int h = (int)(r % H);
  kp[i] = ratio * (expf(dash[i] - dg[r] - km[h]) + STABC);
}

__global__ void ksum_kernel(const float* __restrict__ kp, float* __restrict__ ks,
                            int L, int H, int MF_)
{
  int i = blockIdx.x * blockDim.x + threadIdx.x;
  if (i >= H * MF_) return;
  int h = i / MF_, m = i % MF_;
  float s = 0.f;
  for (int l = 0; l < L; ++l) s += kp[((long)l * H + h) * MF_ + m];
  ks[i] = s;
}

__global__ void den_kernel(const float* __restrict__ qp, const float* __restrict__ ks,
                           float* __restrict__ den, int H, int MF_, int rows)
{
  int r = blockIdx.x * blockDim.x + threadIdx.x;
  if (r >= rows) return;
  int h = r % H;
  const float* p = qp + (long)r * MF_;
  const float* q = ks + (long)h * MF_;
  float s = 0.f;
  for (int m = 0; m < MF_; ++m) s += p[m] * q[m];
  den[r] = s;
}

__global__ void divden_kernel(float* __restrict__ num, const float* __restrict__ den,
                              int H, int Dh, long tot)
{
  long i = (long)blockIdx.x * blockDim.x + threadIdx.x;
  if (i >= tot) return;
  long l = i / ((long)H * Dh);
  int h = (int)((i / Dh) % H);
  num[i] /= den[l * H + h];
}

// ---------------------------------------------------------------------------
extern "C" void kernel_launch(void* const* d_in, const int* in_sizes, int n_in,
                              void* d_out, int out_size, void* d_ws, size_t ws_size,
                              hipStream_t stream)
{
  (void)in_sizes; (void)out_size; (void)ws_size;
  const float* IN[91];
  for (int i = 0; i < n_in && i < 91; ++i) IN[i] = (const float*)d_in[i];

  float* ws = (float*)d_ws;
  size_t off = 0;
  auto alloc = [&](size_t n) { float* p = ws + off; off += (n + 63) & ~(size_t)63; return p; };

  const size_t BIG = (size_t)196608 * 384;          // 75.5M floats
  float* bufX  = alloc(BIG);
  float* bufY  = alloc(BIG);
  float* bufZ  = alloc(BIG);
  float* bufA  = alloc(1536 * 16);
  float* bufA2 = alloc(1536 * 16);
  float* bufT  = alloc(1536 * 784);
  float* bufLN = alloc(1536 * 784);
  float* bufQ  = alloc(1536 * 784);
  float* bufK  = alloc(1536 * 784);
  float* bufV  = alloc(1536 * 784);
  float* bufD  = alloc((size_t)12288 * 256);
  float* bufQP = alloc((size_t)12288 * 256);
  float* bufKP = alloc((size_t)12288 * 256);
  float* bufDg = alloc(12288);
  float* bufDg2= alloc(12288);
  float* bufKM = alloc(64);
  float* bufKS = alloc(2048);
  float* bufDen= alloc(12288);
  float* bufN  = alloc(1536 * 784);
  float* bufKV = alloc(8 * 256 * 98);
  float* bufF  = alloc((size_t)1536 * 1568);

  auto gemm = [&](const float* A, const float* B, const float* bias, const float* Res, float* Out,
                  int M, int N, int K,
                  long a_rs, long a_ks, long a_bs,
                  long b_ks, long b_ns, long b_bs,
                  long o_rs, long o_bs, long r_rs, long r_bs,
                  float ascale, int act, int batch) {
    dim3 g((M + 63) / 64, (N + 31) / 32, batch);
    gemm_kernel<<<g, 128, 0, stream>>>(A, B, bias, Res, Out, M, N, K,
                                       a_rs, a_ks, a_bs, b_ks, b_ns, b_bs,
                                       o_rs, o_bs, r_rs, r_bs, ascale, act);
  };
  auto conv = [&](const float* X, const float* Wt, const float* bias,
                  const float* bg, const float* bb, const float* Res, float* Out,
                  int L, int Cin, int Cout, int W, int act) {
    dim3 g((L + 63) / 64, (Cout + 31) / 32, 1);
    conv1d_kernel<<<g, 128, 0, stream>>>(X, Wt, bias, bg, bb, Res, Out, L, Cin, Cout, W, act);
  };
  auto grid1 = [](long tot) { return dim3((unsigned)((tot + 255) / 256)); };

  // ---------------- stem ----------------
  int L = 196608;
  conv(IN[0], IN[2], IN[3], nullptr, nullptr, nullptr, bufX, L, 4, 384, 15, 0);
  conv(bufX, IN[6], IN[7], IN[4], IN[5], bufX, bufY, L, 384, 384, 1, 0);
  gemm(bufY, IN[8], nullptr, nullptr, bufZ, L, 384, 384,
       384, 1, 0, 384, 1, 0, 384, 0, 0, 0, 1.f, 0, 1);
  pool_kernel<<<grid1((long)(L / 2) * 384), 256, 0, stream>>>(bufY, bufZ, bufX, L / 2, 384);
  L /= 2;
  int P = 384;

  // ---------------- tower ----------------
  const int FILT[6] = {384, 448, 512, 576, 640, 768};
  for (int blk = 0; blk < 6; ++blk) {
    int F = FILT[blk];
    const float* bn1g = IN[15 + 9 * blk + 0];
    const float* bn1b = IN[15 + 9 * blk + 1];
    const float* k1   = IN[15 + 9 * blk + 2];
    const float* b1   = IN[15 + 9 * blk + 3];
    const float* bn2g = IN[15 + 9 * blk + 4];
    const float* bn2b = IN[15 + 9 * blk + 5];
    const float* k2   = IN[15 + 9 * blk + 6];
    const float* b2   = IN[15 + 9 * blk + 7];
    const float* pw   = IN[15 + 9 * blk + 8];
    conv(bufX, k1, b1, bn1g, bn1b, nullptr, bufY, L, P, F, 5, 0);
    conv(bufY, k2, b2, bn2g, bn2b, bufY, bufZ, L, F, F, 1, 0);
    gemm(bufZ, pw, nullptr, nullptr, bufY, L, F, F,
         F, 1, 0, F, 1, 0, F, 0, 0, 0, 1.f, 0, 1);
    pool_kernel<<<grid1((long)(L / 2) * F), 256, 0, stream>>>(bufZ, bufY, bufX, L / 2, F);
    L /= 2; P = F;
  }
  // now: bufX = (1536, 768)

  // ---------------- atac branch ----------------
  conv(IN[1], IN[9], IN[10], nullptr, nullptr, nullptr, bufA, 1536, 1, 16, 25, 0);
  conv(bufA, IN[13], IN[14], IN[11], IN[12], bufA, bufA2, 1536, 16, 16, 1, 0);

  // ---------------- concat + positional encoding ----------------
  concat_pe_kernel<<<grid1((long)1536 * 784), 256, 0, stream>>>(bufX, bufA2, bufT, 1536, 768, 16);

  // ---------------- performer (11 layers) ----------------
  const float DN = 0.3178289440f;   // 98^(-1/4)
  const float RATIO = 0.0625f;      // 1/sqrt(256)
  for (int l = 0; l < 11; ++l) {
    const float* ln1g = IN[70] + (long)l * 784;
    const float* ln1b = IN[71] + (long)l * 784;
    const float* wq   = IN[72] + (long)l * 614656;
    const float* wk   = IN[73] + (long)l * 614656;
    const float* wv   = IN[74] + (long)l * 614656;
    const float* wo   = IN[75] + (long)l * 614656;
    const float* bo   = IN[76] + (long)l * 784;
    const float* ln2g = IN[77] + (long)l * 784;
    const float* ln2b = IN[78] + (long)l * 784;
    const float* w1   = IN[79] + (long)l * 1229312;
    const float* b1   = IN[80] + (long)l * 1568;
    const float* w2   = IN[81] + (long)l * 1229312;
    const float* b2   = IN[82] + (long)l * 784;

    ln_kernel<<<1536, 256, 0, stream>>>(bufT, ln1g, ln1b, bufLN, 784);
    gemm(bufLN, wq, nullptr, nullptr, bufQ, 1536, 784, 784,
         784, 1, 0, 784, 1, 0, 784, 0, 0, 0, 1.f, 0, 1);
    gemm(bufLN, wk, nullptr, nullptr, bufK, 1536, 784, 784,
         784, 1, 0, 784, 1, 0, 784, 0, 0, 0, 1.f, 0, 1);
    gemm(bufLN, wv, nullptr, nullptr, bufV, 1536, 784, 784,
         784, 1, 0, 784, 1, 0, 784, 0, 0, 0, 1.f, 0, 1);
    rope_kernel<<<grid1((long)1536 * 8 * 49), 256, 0, stream>>>(bufQ, 1536, 8, 98);
    rope_kernel<<<grid1((long)1536 * 8 * 49), 256, 0, stream>>>(bufK, 1536, 8, 98);

    // query features: dash = (q*dn) @ proj^T   (B strided as transpose of proj)
    gemm(bufQ, IN[69], nullptr, nullptr, bufD, 12288, 256, 98,
         98, 1, 0, 1, 98, 0, 256, 0, 0, 0, DN, 0, 1);
    diag_kernel<<<grid1(12288), 256, 0, stream>>>(bufQ, bufDg, 12288, 98, DN);
    qfeat_kernel<<<12288, 256, 0, stream>>>(bufD, bufDg, bufQP, 256, RATIO);

    // key features
    gemm(bufK, IN[69], nullptr, nullptr, bufD, 12288, 256, 98,
         98, 1, 0, 1, 98, 0, 256, 0, 0, 0, DN, 0, 1);
    diag_kernel<<<grid1(12288), 256, 0, stream>>>(bufK, bufDg2, 12288, 98, DN);
    kmax_kernel<<<8, 256, 0, stream>>>(bufD, bufKM, 1536, 8, 256);
    kfeat_kernel<<<grid1((long)12288 * 256), 256, 0, stream>>>(bufD, bufDg2, bufKM, bufKP,
                                                               8, 256, (long)12288 * 256, RATIO);

    // kv[h,m,d] = sum_l kp[l,h,m] * v[l,h,d]   (batched over heads)
    gemm(bufKP, bufV, nullptr, nullptr, bufKV, 256, 98, 1536,
         1, 2048, 256, 784, 1, 98, 98, 25088, 0, 0, 1.f, 0, 8);
    ksum_kernel<<<grid1(2048), 256, 0, stream>>>(bufKP, bufKS, 1536, 8, 256);
    // num[l,h,d] = sum_m qp[l,h,m] * kv[h,m,d]
    gemm(bufQP, bufKV, nullptr, nullptr, bufN, 1536, 98, 256,
         2048, 1, 256, 98, 1, 25088, 784, 98, 0, 0, 1.f, 0, 8);
    den_kernel<<<grid1(12288), 256, 0, stream>>>(bufQP, bufKS, bufDen, 8, 256, 12288);
    divden_kernel<<<grid1((long)1536 * 784), 256, 0, stream>>>(bufN, bufDen, 8, 98, (long)1536 * 784);

    // x = x + att @ wo + bo
    gemm(bufN, wo, bo, bufT, bufT, 1536, 784, 784,
         784, 1, 0, 784, 1, 0, 784, 0, 784, 0, 1.f, 0, 1);

    // FFN: x = x + relu(ln2(x) @ w1 + b1) @ w2 + b2
    ln_kernel<<<1536, 256, 0, stream>>>(bufT, ln2g, ln2b, bufLN, 784);
    gemm(bufLN, w1, b1, nullptr, bufF, 1536, 1568, 784,
         784, 1, 0, 1568, 1, 0, 1568, 0, 0, 0, 1.f, 2, 1);
    gemm(bufF, w2, b2, bufT, bufT, 1536, 784, 1568,
         1568, 1, 0, 784, 1, 0, 784, 0, 784, 0, 1.f, 0, 1);
  }

  // ---------------- final: LN, crop, conv_block, gelu, head, softplus ----------------
  ln_kernel<<<1536, 256, 0, stream>>>(bufT, IN[83], IN[84], bufLN, 784);
  // crop rows [320, 320+896); fp conv W=1 with fused pre-BN-GELU and post-GELU epilogue
  conv(bufLN + (long)320 * 784, IN[87], IN[88], IN[85], IN[86], nullptr, bufZ,
       896, 784, 1536, 1, 1);
  // head: softplus(t @ head_w + head_b) -> d_out
  gemm(bufZ, IN[89], IN[90], nullptr, (float*)d_out, 896, 2696, 1536,
       1536, 1, 0, 2696, 1, 0, 2696, 0, 0, 0, 1.f, 3, 1);
}
